// LRU_46935402611364
// MI455X (gfx1250) — compile-verified, matrix operand order checked by reading, not run on hardware
//
#include <hip/hip_runtime.h>
#include <hip/hip_bf16.h>

// ---------------------------------------------------------------------------
// LRU forward for MI455X (gfx1250): L=4, D=256, T=4096, B=8.
// GEMMs: v_wmma_f32_16x16x32_f16, 128x64 block tile, 32x64 wave tile,
//        double-buffered LDS fed by global_load_async_to_lds_b128 (ASYNCcnt).
// Scan: 3-pass chunked parallel scan (local / carry / fix-up).
// ---------------------------------------------------------------------------

typedef __attribute__((ext_vector_type(16))) _Float16 v16h;
typedef __attribute__((ext_vector_type(8)))  float    v8f;

#define LNUM 4
#define Dh   256
#define Tlen 4096
#define Bb   8
#define Mrow (Bb * Tlen)       // 32768 rows (b*T + t), D contiguous
#define TC   128               // scan chunk length
#define NC   (Tlen / TC)       // 32 chunks
#define BD   (Bb * Dh)         // 2048 scan channels

// ---- workspace layout (bytes, 256-aligned) --------------------------------
#define OFF_Z16   ((size_t)0)                                   // M*D f16
#define OFF_BURE  (OFF_Z16  + (size_t)Mrow * Dh * 2)            // M*D f32
#define OFF_BUIM  (OFF_BURE + (size_t)Mrow * Dh * 4)            // M*D f32
#define OFF_G     (OFF_BUIM + (size_t)Mrow * Dh * 4)            // M*D f32
#define OFF_H16   (OFF_G    + (size_t)Mrow * Dh * 4)            // M*2D f16
#define OFF_WCAT  (OFF_H16  + (size_t)Mrow * 2 * Dh * 2)        // L*768*256 f16
#define OFF_CCAT  (OFF_WCAT + (size_t)LNUM * 3 * Dh * Dh * 2)   // L*256*512 f16
#define OFF_CARRY (OFF_CCAT + (size_t)LNUM * Dh * 2 * Dh * 2)   // NC*BD float2
#define OFF_PREF  (OFF_CARRY + (size_t)NC * BD * 8)             // NC*BD float2

// ---------------------------------------------------------------------------
// Async DMA: global -> LDS, 16 bytes per lane, tracked by ASYNCcnt.
// lds_off is the raw LDS byte address (low 32 bits of the flat pointer:
// per ISA, flat->LDS mapping uses addr[31:0]).
// ---------------------------------------------------------------------------
__device__ __forceinline__ void lru_async_b128(unsigned lds_off,
                                               const void* gptr) {
    unsigned long long ga = (unsigned long long)(size_t)gptr;
    asm volatile("global_load_async_to_lds_b128 %0, %1, off"
                 :: "v"(lds_off), "v"(ga) : "memory");
}

__device__ __forceinline__ void lru_wait_async() {
    asm volatile("s_wait_asynccnt 0x0" ::: "memory");
}

__device__ __forceinline__ unsigned lru_lds_off(const void* p) {
    return (unsigned)(size_t)p;   // flat addr low 32 bits == LDS offset
}

// Stage A tile (128x32 halves) and B tile (64x32 halves) for one K-slab.
__device__ __forceinline__ void lru_stage_tiles(const _Float16* __restrict__ A,
                                                const _Float16* __restrict__ W,
                                                _Float16* lA, _Float16* lB,
                                                int m0, int n0, int kk, int K,
                                                int tid) {
    #pragma unroll
    for (int c = tid; c < 512; c += 128) {       // 512 x 16B = 8KB (A)
        int row = c >> 2, col = (c & 3) * 8;
        lru_async_b128(lru_lds_off(&lA[row * 32 + col]),
                       &A[(size_t)(m0 + row) * K + kk + col]);
    }
    #pragma unroll
    for (int c = tid; c < 256; c += 128) {       // 256 x 16B = 4KB (B)
        int row = c >> 2, col = (c & 3) * 8;
        lru_async_b128(lru_lds_off(&lB[row * 32 + col]),
                       &W[(size_t)(n0 + row) * K + kk + col]);
    }
}

// ---------------------------------------------------------------------------
// WMMA fragment loader from a row-major LDS tile (rows x 32 halves).
// 16-bit A/B layout per CDNA5 ISA (16x32 MxK): lane L holds row (L&15),
// half = L>>4 selects K groups {0-7,16-23} vs {8-15,24-31}. Each lane's data
// is two contiguous 16B runs -> compiler emits paired ds_load_b128.
// ---------------------------------------------------------------------------
__device__ __forceinline__ v16h lru_load_frag(const _Float16* __restrict__ lds,
                                              int rowBase, int lane) {
    const int r = rowBase + (lane & 15);
    const int half = lane >> 4;
    const _Float16* p = lds + r * 32;
    v16h f;
    #pragma unroll
    for (int i = 0; i < 8; ++i) {
        int k0 = (i < 4) ? (2 * i + 8 * half) : (16 + 2 * (i - 4) + 8 * half);
        f[2 * i]     = p[k0];
        f[2 * i + 1] = p[k0 + 1];
    }
    return f;
}

// ---------------------------------------------------------------------------
// Weight pre-conversion: Wcat = [B_re; B_im; Wg] (768xK), Ccat = [C_re|-C_im]
// ---------------------------------------------------------------------------
__global__ void lru_convert_weights(const float* __restrict__ B_re,
                                    const float* __restrict__ B_im,
                                    const float* __restrict__ Wg,
                                    const float* __restrict__ C_re,
                                    const float* __restrict__ C_im,
                                    _Float16* __restrict__ Wcat,
                                    _Float16* __restrict__ Ccat) {
    const int NW = LNUM * 3 * Dh * Dh;   // 786432
    const int NCc = LNUM * Dh * 2 * Dh;  // 524288
    int gid = blockIdx.x * blockDim.x + threadIdx.x;
    if (gid < NW) {
        int l = gid / (3 * Dh * Dh);
        int rem = gid % (3 * Dh * Dh);
        int r = rem / Dh, c = rem % Dh;
        float v;
        if (r < Dh)          v = B_re[l * Dh * Dh + r * Dh + c];
        else if (r < 2 * Dh) v = B_im[l * Dh * Dh + (r - Dh) * Dh + c];
        else                 v = Wg[l * Dh * Dh + (r - 2 * Dh) * Dh + c];
        Wcat[gid] = (_Float16)v;
    } else if (gid < NW + NCc) {
        int g2 = gid - NW;
        int l = g2 / (Dh * 2 * Dh);
        int rem = g2 % (Dh * 2 * Dh);
        int d = rem / (2 * Dh), e2 = rem % (2 * Dh);
        float v = (e2 < Dh) ? C_re[l * Dh * Dh + d * Dh + e2]
                            : -C_im[l * Dh * Dh + d * Dh + (e2 - Dh)];
        Ccat[g2] = (_Float16)v;
    }
}

// ---------------------------------------------------------------------------
// LayerNorm: z16[row,:] = LN(u[row,:]) * scale + bias  (f16 out for WMMA A)
// ---------------------------------------------------------------------------
__global__ void lru_layernorm(const float* __restrict__ u,
                              const float* __restrict__ scale,
                              const float* __restrict__ bias,
                              _Float16* __restrict__ z16) {
    __shared__ float s1[Dh];
    __shared__ float s2[Dh];
    const int row = blockIdx.x;
    const int d = threadIdx.x;
    float v = u[(size_t)row * Dh + d];
    s1[d] = v;
    s2[d] = v * v;
    __syncthreads();
    #pragma unroll
    for (int off = Dh / 2; off > 0; off >>= 1) {
        if (d < off) { s1[d] += s1[d + off]; s2[d] += s2[d + off]; }
        __syncthreads();
    }
    float mu = s1[0] * (1.0f / Dh);
    float var = s2[0] * (1.0f / Dh) - mu * mu;
    float z = (v - mu) * rsqrtf(var + 1e-5f) * scale[d] + bias[d];
    z16[(size_t)row * Dh + d] = (_Float16)z;
}

// ---------------------------------------------------------------------------
// GEMM1: out(M x 768) = z16(M x 256) * Wcat(768 x 256)^T
// n<256 -> bu_re*gamma, n<512 -> bu_im*gamma, else gate logits.
// Block: 128 threads (4 waves), block tile 128(M) x 64(N), wave tile 32x64.
// ---------------------------------------------------------------------------
__global__ void lru_gemm_in(const _Float16* __restrict__ A,
                            const _Float16* __restrict__ W,
                            const float* __restrict__ gamma_log,
                            float* __restrict__ buRe,
                            float* __restrict__ buIm,
                            float* __restrict__ gbuf) {
    constexpr int K = Dh;  // 256
    __shared__ _Float16 lA[2][128 * 32];
    __shared__ _Float16 lB[2][64 * 32];
    const int tid = threadIdx.x;
    const int wave = tid >> 5;
    const int lane = tid & 31;
    const int m0 = blockIdx.x * 128;
    const int n0 = blockIdx.y * 64;

    v8f acc[2][4] = {};
    lru_stage_tiles(A, W, lA[0], lB[0], m0, n0, 0, K, tid);
    lru_wait_async();
    __syncthreads();

    int cur = 0;
    for (int kk = 0; kk < K; kk += 32) {
        if (kk + 32 < K)   // prefetch next K-slab while computing this one
            lru_stage_tiles(A, W, lA[cur ^ 1], lB[cur ^ 1], m0, n0, kk + 32, K, tid);
        v16h af0 = lru_load_frag(lA[cur], 32 * wave, lane);
        v16h af1 = lru_load_frag(lA[cur], 32 * wave + 16, lane);
        #pragma unroll
        for (int j = 0; j < 4; ++j) {
            v16h bf = lru_load_frag(lB[cur], 16 * j, lane);
            acc[0][j] = __builtin_amdgcn_wmma_f32_16x16x32_f16(
                false, af0, false, bf, (short)0, acc[0][j], false, false);
            acc[1][j] = __builtin_amdgcn_wmma_f32_16x16x32_f16(
                false, af1, false, bf, (short)0, acc[1][j], false, false);
        }
        lru_wait_async();
        __syncthreads();
        cur ^= 1;
    }

    const int nl = lane & 15, hi = lane >> 4;
    #pragma unroll
    for (int i = 0; i < 2; ++i) {
        #pragma unroll
        for (int j = 0; j < 4; ++j) {
            int n = n0 + 16 * j + nl;
            float gam = (n < 2 * Dh) ? expf(gamma_log[n & (Dh - 1)]) : 1.0f;
            #pragma unroll
            for (int r = 0; r < 8; ++r) {
                int m = m0 + 32 * wave + 16 * i + r + 8 * hi;
                float v = acc[i][j][r];
                if (n < Dh)            buRe[(size_t)m * Dh + n] = v * gam;
                else if (n < 2 * Dh)   buIm[(size_t)m * Dh + (n - Dh)] = v * gam;
                else                   gbuf[(size_t)m * Dh + (n - 2 * Dh)] = v;
            }
        }
    }
}

// ---------------------------------------------------------------------------
// Chunked parallel scan: h[t] = lam*h[t-1] + bu[t], lam complex diagonal.
// ---------------------------------------------------------------------------
__global__ void lru_scan_local(float* __restrict__ buRe, float* __restrict__ buIm,
                               const float* __restrict__ nu_log,
                               const float* __restrict__ theta_log,
                               float2* __restrict__ carry) {
    int gid = blockIdx.x * blockDim.x + threadIdx.x;  // BD*NC = 65536
    int ch = gid % BD, chunk = gid / BD;
    int b = ch / Dh, d = ch % Dh;
    float nu = expf(nu_log[d]), th = expf(theta_log[d]);
    float mag = expf(-nu);
    float lr = mag * __cosf(th), li = mag * __sinf(th);
    float hr = 0.f, hi2 = 0.f;
    size_t base = ((size_t)b * Tlen + (size_t)chunk * TC) * Dh + d;
    for (int tl = 0; tl < TC; ++tl) {
        size_t idx = base + (size_t)tl * Dh;
        float br = buRe[idx], bi = buIm[idx];
        float nr = lr * hr - li * hi2 + br;
        float ni = lr * hi2 + li * hr + bi;
        hr = nr; hi2 = ni;
        buRe[idx] = hr; buIm[idx] = hi2;  // store local (zero-start) scan
    }
    carry[(size_t)chunk * BD + ch] = make_float2(hr, hi2);
}

__global__ void lru_scan_carry(const float2* __restrict__ carry,
                               float2* __restrict__ prefix,
                               const float* __restrict__ nu_log,
                               const float* __restrict__ theta_log) {
    int ch = blockIdx.x * blockDim.x + threadIdx.x;  // BD = 2048
    int d = ch % Dh;
    float nu = expf(nu_log[d]), th = expf(theta_log[d]);
    float mag = expf(-nu * (float)TC);               // lam^TC in closed form
    float ang = th * (float)TC;
    float Lr = mag * __cosf(ang), Li = mag * __sinf(ang);
    float pr = 0.f, pi2 = 0.f;
    #pragma unroll
    for (int c = 0; c < NC; ++c) {
        prefix[(size_t)c * BD + ch] = make_float2(pr, pi2);
        float2 cv = carry[(size_t)c * BD + ch];
        float nr = Lr * pr - Li * pi2 + cv.x;
        float ni = Lr * pi2 + Li * pr + cv.y;
        pr = nr; pi2 = ni;
    }
}

__global__ void lru_scan_fix(const float* __restrict__ buRe,
                             const float* __restrict__ buIm,
                             const float2* __restrict__ prefix,
                             const float* __restrict__ nu_log,
                             const float* __restrict__ theta_log,
                             _Float16* __restrict__ h16) {
    int gid = blockIdx.x * blockDim.x + threadIdx.x;  // BD*NC
    int ch = gid % BD, chunk = gid / BD;
    int b = ch / Dh, d = ch % Dh;
    float nu = expf(nu_log[d]), th = expf(theta_log[d]);
    float mag = expf(-nu);
    float lr = mag * __cosf(th), li = mag * __sinf(th);
    float2 p = prefix[(size_t)chunk * BD + ch];
    float qr = p.x, qi = p.y;
    size_t base = ((size_t)b * Tlen + (size_t)chunk * TC) * Dh + d;
    for (int tl = 0; tl < TC; ++tl) {
        float nqr = lr * qr - li * qi;     // lam^(tl+1) * prefix
        float nqi = lr * qi + li * qr;
        qr = nqr; qi = nqi;
        size_t idx = base + (size_t)tl * Dh;
        float hr = buRe[idx] + qr;
        float hi2 = buIm[idx] + qi;
        size_t row = (size_t)b * Tlen + (size_t)chunk * TC + tl;
        h16[row * (2 * Dh) + d]      = (_Float16)hr;   // A = [h_re | h_im]
        h16[row * (2 * Dh) + Dh + d] = (_Float16)hi2;
    }
}

// ---------------------------------------------------------------------------
// GEMM2: y = [h_re|h_im](M x 512) * [C_re|-C_im](256 x 512)^T
// epilogue: y = (acc + D_vec*z) * sigmoid(g);  u += y  (in place on d_out)
// ---------------------------------------------------------------------------
__global__ void lru_gemm_out(const _Float16* __restrict__ A,
                             const _Float16* __restrict__ W,
                             const float* __restrict__ D_vec,
                             const _Float16* __restrict__ z16,
                             const float* __restrict__ gbuf,
                             float* __restrict__ out) {
    constexpr int K = 2 * Dh;  // 512
    __shared__ _Float16 lA[2][128 * 32];
    __shared__ _Float16 lB[2][64 * 32];
    const int tid = threadIdx.x;
    const int wave = tid >> 5;
    const int lane = tid & 31;
    const int m0 = blockIdx.x * 128;
    const int n0 = blockIdx.y * 64;

    v8f acc[2][4] = {};
    lru_stage_tiles(A, W, lA[0], lB[0], m0, n0, 0, K, tid);
    lru_wait_async();
    __syncthreads();

    int cur = 0;
    for (int kk = 0; kk < K; kk += 32) {
        if (kk + 32 < K)
            lru_stage_tiles(A, W, lA[cur ^ 1], lB[cur ^ 1], m0, n0, kk + 32, K, tid);
        v16h af0 = lru_load_frag(lA[cur], 32 * wave, lane);
        v16h af1 = lru_load_frag(lA[cur], 32 * wave + 16, lane);
        #pragma unroll
        for (int j = 0; j < 4; ++j) {
            v16h bf = lru_load_frag(lB[cur], 16 * j, lane);
            acc[0][j] = __builtin_amdgcn_wmma_f32_16x16x32_f16(
                false, af0, false, bf, (short)0, acc[0][j], false, false);
            acc[1][j] = __builtin_amdgcn_wmma_f32_16x16x32_f16(
                false, af1, false, bf, (short)0, acc[1][j], false, false);
        }
        lru_wait_async();
        __syncthreads();
        cur ^= 1;
    }

    const int nl = lane & 15, hi = lane >> 4;
    #pragma unroll
    for (int i = 0; i < 2; ++i) {
        #pragma unroll
        for (int j = 0; j < 4; ++j) {
            int n = n0 + 16 * j + nl;          // output channel d
            float dv = D_vec[n];
            #pragma unroll
            for (int r = 0; r < 8; ++r) {
                int m = m0 + 32 * wave + 16 * i + r + 8 * hi;
                size_t idx = (size_t)m * Dh + n;
                float z = (float)z16[idx];
                float gate = 1.0f / (1.0f + expf(-gbuf[idx]));
                float y = (acc[i][j][r] + dv * z) * gate;
                out[idx] += y;
            }
        }
    }
}

// ---------------------------------------------------------------------------
extern "C" void kernel_launch(void* const* d_in, const int* in_sizes, int n_in,
                              void* d_out, int out_size, void* d_ws, size_t ws_size,
                              hipStream_t stream) {
    const float* x         = (const float*)d_in[0];
    const float* nu_log    = (const float*)d_in[1];
    const float* theta_log = (const float*)d_in[2];
    const float* gamma_log = (const float*)d_in[3];
    const float* B_re      = (const float*)d_in[4];
    const float* B_im      = (const float*)d_in[5];
    const float* C_re      = (const float*)d_in[6];
    const float* C_im      = (const float*)d_in[7];
    const float* D_vec     = (const float*)d_in[8];
    const float* Wg        = (const float*)d_in[9];
    const float* ln_scale  = (const float*)d_in[10];
    const float* ln_bias   = (const float*)d_in[11];

    char* ws = (char*)d_ws;
    _Float16* z16   = (_Float16*)(ws + OFF_Z16);
    float*    buRe  = (float*)(ws + OFF_BURE);
    float*    buIm  = (float*)(ws + OFF_BUIM);
    float*    gbuf  = (float*)(ws + OFF_G);
    _Float16* h16   = (_Float16*)(ws + OFF_H16);
    _Float16* Wcat  = (_Float16*)(ws + OFF_WCAT);
    _Float16* Ccat  = (_Float16*)(ws + OFF_CCAT);
    float2*   carry = (float2*)(ws + OFF_CARRY);
    float2*   pref  = (float2*)(ws + OFF_PREF);

    float* out = (float*)d_out;

    // residual stream u lives in d_out; seed with x
    hipMemcpyAsync(out, x, (size_t)Mrow * Dh * sizeof(float),
                   hipMemcpyDeviceToDevice, stream);

    // convert all layer weights to f16 concat blocks once
    {
        int total = LNUM * 3 * Dh * Dh + LNUM * Dh * 2 * Dh;
        lru_convert_weights<<<(total + 255) / 256, 256, 0, stream>>>(
            B_re, B_im, Wg, C_re, C_im, Wcat, Ccat);
    }

    for (int l = 0; l < LNUM; ++l) {
        lru_layernorm<<<Mrow, Dh, 0, stream>>>(
            out, ln_scale + l * Dh, ln_bias + l * Dh, z16);

        lru_gemm_in<<<dim3(Mrow / 128, (3 * Dh) / 64), 128, 0, stream>>>(
            z16, Wcat + (size_t)l * 3 * Dh * Dh, gamma_log + l * Dh,
            buRe, buIm, gbuf);

        lru_scan_local<<<(BD * NC) / 256, 256, 0, stream>>>(
            buRe, buIm, nu_log + l * Dh, theta_log + l * Dh, carry);
        lru_scan_carry<<<BD / 256, 256, 0, stream>>>(
            carry, pref, nu_log + l * Dh, theta_log + l * Dh);
        lru_scan_fix<<<(BD * NC) / 256, 256, 0, stream>>>(
            buRe, buIm, pref, nu_log + l * Dh, theta_log + l * Dh, h16);

        lru_gemm_out<<<dim3(Mrow / 128, Dh / 64), 128, 0, stream>>>(
            h16, Ccat + (size_t)l * Dh * 2 * Dh, D_vec + l * Dh,
            z16, gbuf, out);
    }
}
